// Peehole_45526653337954
// MI455X (gfx1250) — matrix operand order repaired and checked by reading, not compiled, not run on hardware
//
#include <hip/hip_runtime.h>
#include <hip/hip_bf16.h>

// ---------------------------------------------------------------------------
// Peephole LSTM on gfx1250 (MI455X): bf16 WMMA GEMMs, fp32 accumulation.
//   I = H = 1024, B = 64, T = 256
// Phase 1: pregate GEMM  X[16384,1024] x W_i2{i,f,o,g}^T  -> G[4][T][B][H] f32
// Phase 2: 256 sequential steps, each split into:
//   A: i,f,g gates (h@Wh2*, c@Wc2* GEMMs) -> c_new
//   B: o gate (h@Wh2o, c_new@Wc2o GEMMs)  -> h_new
// K-loops are software-pipelined (prefetch next fragments before WMMA) so the
// s_wait_loadcnt before each v_wmma covers loads issued one iteration ago.
// ---------------------------------------------------------------------------

typedef __bf16 bf16_t;
typedef __attribute__((ext_vector_type(16))) __bf16 v16bf;
typedef __attribute__((ext_vector_type(8)))  __bf16 v8bf;
typedef __attribute__((ext_vector_type(8)))  float  v8f;

#define IDIM 1024
#define HDIM 1024
#define BDIM 64
#define TDIM 256

// ---------------- WMMA fragment helpers (contiguous k order, same in A and B) --------

// A fragment: 16x32 bf16 tile, row-major source with stride lda.
// lanes 0-15: M=0..15, K = 0..15
// lanes16-31: M=0..15, K = 16..31
__device__ __forceinline__ v16bf load_a_frag(const bf16_t* __restrict__ base,
                                             int m0, int k0, int lda, int lane) {
    int m  = m0 + (lane & 15);
    int kb = (lane >> 4) << 4;               // 0 or 16
    const bf16_t* p = base + (size_t)m * lda + k0 + kb;
    return *(const v16bf*)(p);               // K = kb .. kb+15
}

// B fragment: 32x16 tile of B = W^T where W is row-major [N=H, K].
// lane n = lane&15; lanes 0-15 hold K=0..15, lanes 16-31 hold K=16..31
// -> a single contiguous 32-byte load along W's row.
__device__ __forceinline__ v16bf load_b_frag(const bf16_t* __restrict__ w,
                                             int n0, int k0, int ldk, int lane) {
    int n  = n0 + (lane & 15);
    int kb = (lane >> 4) << 4;               // 0 or 16
    return *(const v16bf*)(w + (size_t)n * ldk + k0 + kb);
}

__device__ __forceinline__ v8f wmma_bf16(v16bf a, v16bf b, v8f c) {
    return __builtin_amdgcn_wmma_f32_16x16x32_bf16(false, a, false, b,
                                                   (short)0, c, false, false);
}

__device__ __forceinline__ float sigmoidf_(float x) {
    return 1.0f / (1.0f + __expf(-x));
}

// ---------------- elementwise conversion / init -----------------------------

__global__ void f32_to_bf16_kernel(const float* __restrict__ src,
                                   bf16_t* __restrict__ dst, int n) {
    int i = blockIdx.x * blockDim.x + threadIdx.x;
    if (i < n) dst[i] = (bf16_t)src[i];
}

__global__ void init_state_kernel(bf16_t* __restrict__ hbf,
                                  bf16_t* __restrict__ cbf,
                                  float* __restrict__ cf, int n) {
    int i = blockIdx.x * blockDim.x + threadIdx.x;
    if (i < n) { hbf[i] = (bf16_t)0.0f; cbf[i] = (bf16_t)0.0f; cf[i] = 0.0f; }
}

// ---------------- Phase 1: input->gate projections --------------------------
// Tasks: gate(4) x rowtile(1024 of 16 rows) x ngroup(16 of 64 cols), 4 waves/block.
// Each wave: 16x64 output, pipelined K-loop over 1024.
__global__ void pregate_kernel(const bf16_t* __restrict__ Xbf,   // [B*T, I]
                               const bf16_t* __restrict__ Wbf,   // 4 gates, H*I each
                               const float* __restrict__ b0, const float* __restrict__ b1,
                               const float* __restrict__ b2, const float* __restrict__ b3,
                               float* __restrict__ G) {          // [4][T][B][H]
    int lane = threadIdx.x & 31;
    int wave = threadIdx.x >> 5;
    int task = blockIdx.x * (blockDim.x >> 5) + wave;
    int gate = task >> 14;                    // / (1024*16)
    int rem  = task & 16383;
    int m0 = (rem >> 4) << 4;                 // row tile base in X
    int n0 = (rem & 15) << 6;                 // 64-wide N group
    const bf16_t* W = Wbf + (size_t)gate * HDIM * IDIM;
    const float* bias = (gate == 0) ? b0 : (gate == 1) ? b1 : (gate == 2) ? b2 : b3;

    v8f acc[4] = {};
    // pipeline: prefetch fragments for the next K step before consuming current
    v16bf a_c = load_a_frag(Xbf, m0, 0, IDIM, lane);
    v16bf b_c[4];
#pragma unroll
    for (int j = 0; j < 4; ++j) b_c[j] = load_b_frag(W, n0 + j * 16, 0, IDIM, lane);

    for (int k0 = 0; k0 < IDIM; k0 += 32) {
        int kn = (k0 + 32 < IDIM) ? k0 + 32 : 0;   // clamp: harmless redundant load
        v16bf a_n = load_a_frag(Xbf, m0, kn, IDIM, lane);
        v16bf b_n[4];
#pragma unroll
        for (int j = 0; j < 4; ++j) b_n[j] = load_b_frag(W, n0 + j * 16, kn, IDIM, lane);
#pragma unroll
        for (int j = 0; j < 4; ++j) acc[j] = wmma_bf16(a_c, b_c[j], acc[j]);
        a_c = a_n;
#pragma unroll
        for (int j = 0; j < 4; ++j) b_c[j] = b_n[j];
    }

    float* Gg = G + (size_t)gate * TDIM * BDIM * HDIM;
    int col   = lane & 15;
    int rbase = (lane >> 4) << 3;
#pragma unroll
    for (int j = 0; j < 4; ++j) {
        int n = n0 + j * 16 + col;
        float bv = bias[n];
#pragma unroll
        for (int r = 0; r < 8; ++r) {
            int xrow = m0 + rbase + r;        // = b*T + t
            int bidx = xrow >> 8;
            int tidx = xrow & 255;
            Gg[((size_t)tidx * BDIM + bidx) * HDIM + n] = acc[j][r] + bv;
        }
    }
}

// ---------------- Phase 2A: i,f,g gates + c_new ------------------------------
// 256 tile tasks (4 m-tiles x 64 n-tiles), 4 waves/block -> 64 blocks.
__global__ void lstm_stepA(const float* __restrict__ Gi, const float* __restrict__ Gf,
                           const float* __restrict__ Gg,
                           const bf16_t* __restrict__ hbf, const bf16_t* __restrict__ cbf,
                           const float* __restrict__ cf32,
                           const bf16_t* __restrict__ Whi, const bf16_t* __restrict__ Wci,
                           const bf16_t* __restrict__ Whf, const bf16_t* __restrict__ Wcf,
                           const bf16_t* __restrict__ Whg,
                           const float* __restrict__ bhi, const float* __restrict__ bci,
                           const float* __restrict__ bhf, const float* __restrict__ bcf,
                           const float* __restrict__ bhg,
                           float* __restrict__ cf32_out, bf16_t* __restrict__ cbf_out,
                           float* __restrict__ cT_out) {
    int lane = threadIdx.x & 31;
    int wave = threadIdx.x >> 5;
    int task = blockIdx.x * (blockDim.x >> 5) + wave;   // 0..255
    int m0 = (task >> 6) << 4;
    int n0 = (task & 63) << 4;

    v8f ai = {}, af = {}, ag = {};

    v16bf ah_c = load_a_frag(hbf, m0, 0, HDIM, lane);
    v16bf ac_c = load_a_frag(cbf, m0, 0, HDIM, lane);
    v16bf bhi_c = load_b_frag(Whi, n0, 0, HDIM, lane);
    v16bf bci_c = load_b_frag(Wci, n0, 0, HDIM, lane);
    v16bf bhf_c = load_b_frag(Whf, n0, 0, HDIM, lane);
    v16bf bcf_c = load_b_frag(Wcf, n0, 0, HDIM, lane);
    v16bf bhg_c = load_b_frag(Whg, n0, 0, HDIM, lane);

    for (int k0 = 0; k0 < HDIM; k0 += 32) {
        int kn = (k0 + 32 < HDIM) ? k0 + 32 : 0;
        v16bf ah_n  = load_a_frag(hbf, m0, kn, HDIM, lane);
        v16bf ac_n  = load_a_frag(cbf, m0, kn, HDIM, lane);
        v16bf bhi_n = load_b_frag(Whi, n0, kn, HDIM, lane);
        v16bf bci_n = load_b_frag(Wci, n0, kn, HDIM, lane);
        v16bf bhf_n = load_b_frag(Whf, n0, kn, HDIM, lane);
        v16bf bcf_n = load_b_frag(Wcf, n0, kn, HDIM, lane);
        v16bf bhg_n = load_b_frag(Whg, n0, kn, HDIM, lane);

        ai = wmma_bf16(ah_c, bhi_c, ai);
        ai = wmma_bf16(ac_c, bci_c, ai);
        af = wmma_bf16(ah_c, bhf_c, af);
        af = wmma_bf16(ac_c, bcf_c, af);
        ag = wmma_bf16(ah_c, bhg_c, ag);

        ah_c = ah_n;  ac_c = ac_n;
        bhi_c = bhi_n; bci_c = bci_n; bhf_c = bhf_n; bcf_c = bcf_n; bhg_c = bhg_n;
    }

    int col   = lane & 15;
    int rbase = (lane >> 4) << 3;
    int n = n0 + col;
    float bii = bhi[n] + bci[n];
    float bff = bhf[n] + bcf[n];
    float bgg = bhg[n];
#pragma unroll
    for (int r = 0; r < 8; ++r) {
        int m = m0 + rbase + r;
        size_t idx = (size_t)m * HDIM + n;
        float ig = sigmoidf_(ai[r] + Gi[idx] + bii);
        float fg = sigmoidf_(af[r] + Gf[idx] + bff);
        float gg = tanhf(ag[r] + Gg[idx] + bgg);
        float cn = fg * cf32[idx] + ig * gg;
        cf32_out[idx] = cn;
        cbf_out[idx]  = (bf16_t)cn;
        if (cT_out) cT_out[idx] = cn;
    }
}

// ---------------- Phase 2B: o gate + h_new -----------------------------------
__global__ void lstm_stepB(const float* __restrict__ Go,
                           const bf16_t* __restrict__ hbf, const bf16_t* __restrict__ cnbf,
                           const float* __restrict__ cnf32,
                           const bf16_t* __restrict__ Who, const bf16_t* __restrict__ Wco,
                           const float* __restrict__ bho, const float* __restrict__ bco,
                           bf16_t* __restrict__ hbf_out,
                           float* __restrict__ hidden_out, int t,
                           float* __restrict__ hT_out) {
    int lane = threadIdx.x & 31;
    int wave = threadIdx.x >> 5;
    int task = blockIdx.x * (blockDim.x >> 5) + wave;   // 0..255
    int m0 = (task >> 6) << 4;
    int n0 = (task & 63) << 4;

    v8f ao = {};

    v16bf ah_c = load_a_frag(hbf,  m0, 0, HDIM, lane);
    v16bf an_c = load_a_frag(cnbf, m0, 0, HDIM, lane);
    v16bf bo_c = load_b_frag(Who, n0, 0, HDIM, lane);
    v16bf bc_c = load_b_frag(Wco, n0, 0, HDIM, lane);

    for (int k0 = 0; k0 < HDIM; k0 += 32) {
        int kn = (k0 + 32 < HDIM) ? k0 + 32 : 0;
        v16bf ah_n = load_a_frag(hbf,  m0, kn, HDIM, lane);
        v16bf an_n = load_a_frag(cnbf, m0, kn, HDIM, lane);
        v16bf bo_n = load_b_frag(Who, n0, kn, HDIM, lane);
        v16bf bc_n = load_b_frag(Wco, n0, kn, HDIM, lane);

        ao = wmma_bf16(ah_c, bo_c, ao);
        ao = wmma_bf16(an_c, bc_c, ao);

        ah_c = ah_n; an_c = an_n; bo_c = bo_n; bc_c = bc_n;
    }

    int col   = lane & 15;
    int rbase = (lane >> 4) << 3;
    int n = n0 + col;
    float boo = bho[n] + bco[n];
#pragma unroll
    for (int r = 0; r < 8; ++r) {
        int m = m0 + rbase + r;
        size_t idx = (size_t)m * HDIM + n;
        float og = sigmoidf_(ao[r] + Go[idx] + boo);
        float hn = og * tanhf(cnf32[idx]);
        hbf_out[idx] = (bf16_t)hn;
        hidden_out[((size_t)m * TDIM + t) * HDIM + n] = hn;   // [B,T,H]
        if (hT_out) hT_out[idx] = hn;
    }
}

// ---------------- host launch ------------------------------------------------

extern "C" void kernel_launch(void* const* d_in, const int* in_sizes, int n_in,
                              void* d_out, int out_size, void* d_ws, size_t ws_size,
                              hipStream_t stream) {
    (void)in_sizes; (void)n_in; (void)out_size; (void)ws_size;
    const float* x = (const float*)d_in[0];
    // order: i2i, i2f, i2o, i2g, h2i, h2f, h2o, h2g, c2i, c2f, c2o
    const float* W[11];
    const float* Bv[11];
    for (int j = 0; j < 11; ++j) {
        W[j]  = (const float*)d_in[1 + 2 * j];
        Bv[j] = (const float*)d_in[2 + 2 * j];
    }

    const int    wn = HDIM * IDIM;               // 1M elems per weight matrix
    const size_t stateN = (size_t)BDIM * HDIM;   // 65536

    char* ws = (char*)d_ws;
    size_t off = 0;
    bf16_t* Wbf = (bf16_t*)(ws + off);  off += (size_t)11 * wn * sizeof(bf16_t);
    bf16_t* Xbf = (bf16_t*)(ws + off);  off += (size_t)BDIM * TDIM * IDIM * sizeof(bf16_t);
    float*  G   = (float* )(ws + off);  off += (size_t)4 * TDIM * BDIM * HDIM * sizeof(float);
    float*  cf[2];  cf[0]  = (float* )(ws + off); off += stateN * sizeof(float);
                    cf[1]  = (float* )(ws + off); off += stateN * sizeof(float);
    bf16_t* cbf[2]; cbf[0] = (bf16_t*)(ws + off); off += stateN * sizeof(bf16_t);
                    cbf[1] = (bf16_t*)(ws + off); off += stateN * sizeof(bf16_t);
    bf16_t* hbf[2]; hbf[0] = (bf16_t*)(ws + off); off += stateN * sizeof(bf16_t);
                    hbf[1] = (bf16_t*)(ws + off); off += stateN * sizeof(bf16_t);

    // 1) convert weights and x to bf16
    for (int j = 0; j < 11; ++j)
        f32_to_bf16_kernel<<<wn / 256, 256, 0, stream>>>(W[j], Wbf + (size_t)j * wn, wn);
    const int xn = BDIM * TDIM * IDIM;
    f32_to_bf16_kernel<<<xn / 256, 256, 0, stream>>>(x, Xbf, xn);

    // 2) zero initial state (h0 = c0 = 0)
    init_state_kernel<<<(int)(stateN / 256), 256, 0, stream>>>(hbf[0], cbf[0], cf[0],
                                                               (int)stateN);

    // 3) input->gate projections: 65536 wave tasks, 4 waves/block
    pregate_kernel<<<16384, 128, 0, stream>>>(Xbf, Wbf, Bv[0], Bv[1], Bv[2], Bv[3], G);

    // 4) sequential scan
    float* out = (float*)d_out;
    float* hs  = out;                                   // [B,T,H]
    float* hT  = out + (size_t)BDIM * TDIM * HDIM;      // [B,1,H]
    float* cT  = hT + stateN;                           // [B,1,H]

    const size_t GTB = (size_t)TDIM * BDIM * HDIM;
    const bf16_t* Whi = Wbf + (size_t)4  * wn;
    const bf16_t* Whf = Wbf + (size_t)5  * wn;
    const bf16_t* Who = Wbf + (size_t)6  * wn;
    const bf16_t* Whg = Wbf + (size_t)7  * wn;
    const bf16_t* Wci = Wbf + (size_t)8  * wn;
    const bf16_t* Wcf = Wbf + (size_t)9  * wn;
    const bf16_t* Wco = Wbf + (size_t)10 * wn;

    for (int t = 0; t < TDIM; ++t) {
        int cur = t & 1, nxt = cur ^ 1;
        size_t toff = (size_t)t * BDIM * HDIM;
        const float* Gi = G + 0 * GTB + toff;
        const float* Gf = G + 1 * GTB + toff;
        const float* Go = G + 2 * GTB + toff;
        const float* Gg = G + 3 * GTB + toff;
        bool last = (t == TDIM - 1);

        lstm_stepA<<<64, 128, 0, stream>>>(Gi, Gf, Gg,
            hbf[cur], cbf[cur], cf[cur],
            Whi, Wci, Whf, Wcf, Whg,
            Bv[4], Bv[8], Bv[5], Bv[9], Bv[7],
            cf[nxt], cbf[nxt], last ? cT : nullptr);

        lstm_stepB<<<64, 128, 0, stream>>>(Go,
            hbf[cur], cbf[nxt], cf[nxt],
            Who, Wco, Bv[6], Bv[10],
            hbf[nxt], hs, t, last ? hT : nullptr);
    }
}
